// RNN_Model_88819923681377
// MI455X (gfx1250) — matrix operand order repaired and checked by reading, not compiled, not run on hardware
//
#include <hip/hip_runtime.h>

typedef __attribute__((ext_vector_type(8)))  float   v8f;
typedef __attribute__((ext_vector_type(8)))  __bf16  v8bf;
typedef __attribute__((ext_vector_type(16))) __bf16  v16bf;

#define T_STEPS 300
#define IN_DIM  38
#define H1      32
#define H2      20
#define K1      96   // layer1 K: 38 + 32 + 1(bias) padded to 96 -> 3 k-tiles
#define K2      64   // layer2 K: 32 + 20 + 1(bias) padded to 64 -> 2 k-tiles

#define LDS_FENCE() asm volatile("s_wait_dscnt 0x0" ::: "memory")

__device__ __forceinline__ float sigm(float x) { return 1.0f / (1.0f + __expf(-x)); }
__device__ __forceinline__ float tanh_f(float x) {
  float a = fabsf(x);
  float e = __expf(-2.0f * a);
  float t = (1.0f - e) / (1.0f + e);
  return copysignf(t, x);
}

// A-fragment gather for V_WMMA_*_16X16X32 (16-bit): lane holds row M=lane%16,
// K = { hi*16 .. hi*16+15 } within the 32-wide k-chunk.
__device__ __forceinline__ v16bf ldsA(const __bf16* p, int hi) {
  return *(const v16bf*)(p + hi * 16);
}

// ---------------------------------------------------------------------------
// Bidirectional stacked-LSTM scan. One wave32 per 16-row batch tile per
// direction; whole 300-step recurrence stays in registers + LDS.
// grid = (B/32, 2 directions), block = 64 (2 waves sharing weight LDS).
// ---------------------------------------------------------------------------
__global__ __launch_bounds__(64) void bilstm_scan_kernel(
    const float* __restrict__ x,
    const float* __restrict__ Wf1, const float* __restrict__ bf1,
    const float* __restrict__ Wf2, const float* __restrict__ bf2,
    const float* __restrict__ Wb1, const float* __restrict__ bb1,
    const float* __restrict__ Wb2, const float* __restrict__ bb2,
    float* __restrict__ feat)
{
  __shared__ __align__(32) __bf16 w1p[3 * 8 * 512];   // 24 KB, B-frag layout
  __shared__ __align__(32) __bf16 w2p[2 * 5 * 512];   // 10 KB
  __shared__ __align__(32) __bf16 act1[2][16 * K1];   // [row][feat]: x | h1 | 1 | 0
  __shared__ __align__(32) __bf16 act2[2][16 * K2];   // [row][feat]: h1 | h2 | 1 | 0
  __shared__ float zbuf[2][16 * 80];                  // layer2 pre-activation staging

  const int tid  = threadIdx.x;
  const int wave = tid >> 5;
  const int l    = tid & 31;
  const int hi   = l >> 4;
  const int n0   = l & 15;
  const int dir  = blockIdx.y;

  const float* W1 = dir ? Wb1 : Wf1;
  const float* b1 = dir ? bb1 : bf1;
  const float* W2 = dir ? Wb2 : Wf2;
  const float* b2 = dir ? bb2 : bf2;

  // Pack weights (+ bias as extra K row) into WMMA B-fragment order:
  // element i of lane L in tile (c,t) = W[c*32 + (L/16)*16 + i][t*16 + L%16]
  for (int d = tid; d < 3 * 8 * 512; d += 64) {
    int tile = d >> 9, rem = d & 511;
    int lane = rem >> 4, i = rem & 15;
    int c = tile >> 3, t = tile & 7;
    int k = c * 32 + (lane >> 4) * 16 + i;
    int n = t * 16 + (lane & 15);
    float v = 0.0f;
    if (k < IN_DIM + H1) v = W1[k * 128 + n];
    else if (k == IN_DIM + H1) v = b1[n];
    w1p[d] = (__bf16)v;
  }
  for (int d = tid; d < 2 * 5 * 512; d += 64) {
    int tile = d >> 9, rem = d & 511;
    int lane = rem >> 4, i = rem & 15;
    int c = tile / 5, t = tile % 5;
    int k = c * 32 + (lane >> 4) * 16 + i;
    int n = t * 16 + (lane & 15);
    float v = 0.0f;
    if (k < H1 + H2) v = W2[k * 80 + n];
    else if (k == H1 + H2) v = b2[n];
    w2p[d] = (__bf16)v;
  }

  __bf16* a1w = act1[wave];
  __bf16* a2w = act2[wave];
  // Initial hidden state = 0; bias-one feature; zero K padding.
  for (int d = l; d < 16 * K1; d += 32) {
    int k = d % K1;
    if (k >= IN_DIM) a1w[d] = (__bf16)((k == IN_DIM + H1) ? 1.0f : 0.0f);
  }
  for (int d = l; d < 16 * K2; d += 32) {
    int k = d % K2;
    if (k >= H1) a2w[d] = (__bf16)((k == H1 + H2) ? 1.0f : 0.0f);
  }
  __syncthreads();

  const int tileBase = (blockIdx.x * 2 + wave) * 16;
  float c1s[2][8] = {};   // layer1 cell state: rows hi*8+r, units g*16+n0
  float c2s[10]   = {};   // layer2 cell state: item m = l*10+q -> (m/20, m%20)
  float* zw = zbuf[wave];

  #pragma unroll 1
  for (int s = 0; s < T_STEPS; ++s) {
    const int tmem = dir ? (T_STEPS - 1 - s) : s;

    // Stage x_t (16 x 38) as bf16 into act1. 608 = 32*19 -> uniform trip count.
    for (int idx = l; idx < 16 * IN_DIM; idx += 32) {
      int r = idx / IN_DIM, k = idx - r * IN_DIM;
      a1w[r * K1 + k] = (__bf16)x[((size_t)(tileBase + r) * T_STEPS + tmem) * IN_DIM + k];
    }
    LDS_FENCE();

    // layer1: z1 = [x_t, h1_{t-1}, 1] @ W1  ->  16 x 128 (i|j|f|o, 32 each)
    v8f z1[8] = {};
    #pragma unroll
    for (int c = 0; c < 3; ++c) {
      v16bf a = ldsA(&a1w[n0 * K1 + c * 32], hi);
      #pragma unroll
      for (int t = 0; t < 8; ++t) {
        v16bf bm = *(const v16bf*)&w1p[(c * 8 + t) * 512 + l * 16];
        z1[t] = __builtin_amdgcn_wmma_f32_16x16x32_bf16(false, a, false, bm,
                                                        (short)0, z1[t], false, false);
      }
    }
    // layer1 gates on the C/D fragment layout (row = hi*8+r, unit = g*16+n0)
    #pragma unroll
    for (int g = 0; g < 2; ++g) {
      const int u = g * 16 + n0;
      #pragma unroll
      for (int r = 0; r < 8; ++r) {
        float iv = sigm(z1[g][r]);
        float jv = tanh_f(z1[2 + g][r]);
        float fv = sigm(z1[4 + g][r] + 1.0f);   // forget_bias = 1.0
        float ov = sigm(z1[6 + g][r]);
        float cc = c1s[g][r] * fv + iv * jv;
        c1s[g][r] = cc;
        __bf16 hb = (__bf16)(tanh_f(cc) * ov);
        const int row = hi * 8 + r;
        a1w[row * K1 + IN_DIM + u] = hb;   // recurrence of layer1
        a2w[row * K2 + u]          = hb;   // input to layer2
      }
    }
    LDS_FENCE();

    // layer2: z2 = [h1_t, h2_{t-1}, 1] @ W2  ->  16 x 80 (i|j|f|o, 20 each)
    v8f z2[5] = {};
    #pragma unroll
    for (int c = 0; c < 2; ++c) {
      v16bf a = ldsA(&a2w[n0 * K2 + c * 32], hi);
      #pragma unroll
      for (int t = 0; t < 5; ++t) {
        v16bf bm = *(const v16bf*)&w2p[(c * 5 + t) * 512 + l * 16];
        z2[t] = __builtin_amdgcn_wmma_f32_16x16x32_bf16(false, a, false, bm,
                                                        (short)0, z2[t], false, false);
      }
    }
    // gate stride (20) is not lane-aligned -> stage z2 through LDS (f32)
    #pragma unroll
    for (int t = 0; t < 5; ++t)
      #pragma unroll
      for (int r = 0; r < 8; ++r)
        zw[(hi * 8 + r) * 80 + t * 16 + n0] = z2[t][r];
    LDS_FENCE();

    #pragma unroll
    for (int q = 0; q < 10; ++q) {
      const int m = l * 10 + q;
      const int row = m / 20, u = m - row * 20;
      float iv = sigm(zw[row * 80 + u]);
      float jv = tanh_f(zw[row * 80 + 20 + u]);
      float fv = sigm(zw[row * 80 + 40 + u] + 1.0f);
      float ov = sigm(zw[row * 80 + 60 + u]);
      float cc = c2s[q] * fv + iv * jv;
      c2s[q] = cc;
      float h = tanh_f(cc) * ov;
      a2w[row * K2 + H1 + u] = (__bf16)h;
      // only scan steps 0 and T-1 feed the head:
      // fw: step0 -> feat[0:20],  stepT-1 -> feat[40:60]
      // bw: step0 -> feat[60:80], stepT-1 -> feat[20:40]
      if (s == 0)
        feat[(size_t)(tileBase + row) * 80 + (dir ? 60 : 0) + u] = h;
      if (s == T_STEPS - 1)
        feat[(size_t)(tileBase + row) * 80 + (dir ? 20 : 40) + u] = h;
    }
    LDS_FENCE();
  }
}

// ---------------------------------------------------------------------------
// FC head: feat[B,80] -> tanh(80) -> tanh(20) -> 3. One wave per 16 rows.
// ---------------------------------------------------------------------------
__global__ __launch_bounds__(32) void fc_head_kernel(
    const float* __restrict__ feat,
    const float* __restrict__ W1, const float* __restrict__ b1,
    const float* __restrict__ W2, const float* __restrict__ b2,
    const float* __restrict__ W3, const float* __restrict__ b3,
    float* __restrict__ out)
{
  __shared__ __align__(32) __bf16 w1p[3 * 5 * 512];   // 81x80 padded to 96x80
  __shared__ __align__(32) __bf16 w2p[3 * 2 * 512];   // 81x20 padded to 96x32
  __shared__ __align__(32) __bf16 actf[16 * 96];
  __shared__ __align__(32) __bf16 acth[16 * 96];
  __shared__ float h2s[16 * 20];
  __shared__ float w3s[20 * 3];
  __shared__ float b3s[3];

  const int l = threadIdx.x;
  const int hi = l >> 4, n0 = l & 15;

  for (int d = l; d < 3 * 5 * 512; d += 32) {
    int tile = d >> 9, rem = d & 511;
    int lane = rem >> 4, i = rem & 15;
    int c = tile / 5, t = tile % 5;
    int k = c * 32 + (lane >> 4) * 16 + i;
    int n = t * 16 + (lane & 15);
    float v = 0.0f;
    if (k < 80) v = W1[k * 80 + n];
    else if (k == 80) v = b1[n];
    w1p[d] = (__bf16)v;
  }
  for (int d = l; d < 3 * 2 * 512; d += 32) {
    int tile = d >> 9, rem = d & 511;
    int lane = rem >> 4, i = rem & 15;
    int c = tile >> 1, t = tile & 1;
    int k = c * 32 + (lane >> 4) * 16 + i;
    int n = t * 16 + (lane & 15);
    float v = 0.0f;
    if (n < 20) {
      if (k < 80) v = W2[k * 20 + n];
      else if (k == 80) v = b2[n];
    }
    w2p[d] = (__bf16)v;
  }
  for (int d = l; d < 60; d += 32) w3s[d] = W3[d];
  if (l < 3) b3s[l] = b3[l];

  const int base = blockIdx.x * 16;
  for (int idx = l; idx < 16 * 96; idx += 32) {
    int r = idx / 96, k = idx - r * 96;
    float v = (k < 80) ? feat[(size_t)(base + r) * 80 + k] : ((k == 80) ? 1.0f : 0.0f);
    actf[idx] = (__bf16)v;
    if (k >= 80) acth[idx] = (__bf16)((k == 80) ? 1.0f : 0.0f);
  }
  LDS_FENCE();

  v8f z1[5] = {};
  #pragma unroll
  for (int c = 0; c < 3; ++c) {
    v16bf a = ldsA(&actf[n0 * 96 + c * 32], hi);
    #pragma unroll
    for (int t = 0; t < 5; ++t) {
      v16bf bm = *(const v16bf*)&w1p[(c * 5 + t) * 512 + l * 16];
      z1[t] = __builtin_amdgcn_wmma_f32_16x16x32_bf16(false, a, false, bm,
                                                      (short)0, z1[t], false, false);
    }
  }
  #pragma unroll
  for (int t = 0; t < 5; ++t)
    #pragma unroll
    for (int r = 0; r < 8; ++r)
      acth[(hi * 8 + r) * 96 + t * 16 + n0] = (__bf16)tanh_f(z1[t][r]);
  LDS_FENCE();

  v8f z2[2] = {};
  #pragma unroll
  for (int c = 0; c < 3; ++c) {
    v16bf a = ldsA(&acth[n0 * 96 + c * 32], hi);
    #pragma unroll
    for (int t = 0; t < 2; ++t) {
      v16bf bm = *(const v16bf*)&w2p[(c * 2 + t) * 512 + l * 16];
      z2[t] = __builtin_amdgcn_wmma_f32_16x16x32_bf16(false, a, false, bm,
                                                      (short)0, z2[t], false, false);
    }
  }
  #pragma unroll
  for (int t = 0; t < 2; ++t) {
    const int col = t * 16 + n0;
    if (col < 20) {
      #pragma unroll
      for (int r = 0; r < 8; ++r)
        h2s[(hi * 8 + r) * 20 + col] = tanh_f(z2[t][r]);
    }
  }
  LDS_FENCE();

  if (l < 16) {
    #pragma unroll
    for (int j = 0; j < 3; ++j) {
      float acc = b3s[j];
      #pragma unroll
      for (int u = 0; u < 20; ++u) acc += h2s[l * 20 + u] * w3s[u * 3 + j];
      out[(size_t)(base + l) * 3 + j] = acc;
    }
  }
}

extern "C" void kernel_launch(void* const* d_in, const int* in_sizes, int n_in,
                              void* d_out, int out_size, void* d_ws, size_t ws_size,
                              hipStream_t stream) {
  const float* x    = (const float*)d_in[0];
  const float* Wf1  = (const float*)d_in[1];
  const float* bf1  = (const float*)d_in[2];
  const float* Wf2  = (const float*)d_in[3];
  const float* bf2  = (const float*)d_in[4];
  const float* Wb1  = (const float*)d_in[5];
  const float* bb1  = (const float*)d_in[6];
  const float* Wb2  = (const float*)d_in[7];
  const float* bb2  = (const float*)d_in[8];
  const float* Wfc1 = (const float*)d_in[9];
  const float* bfc1 = (const float*)d_in[10];
  const float* Wfc2 = (const float*)d_in[11];
  const float* bfc2 = (const float*)d_in[12];
  const float* Wfc3 = (const float*)d_in[13];
  const float* bfc3 = (const float*)d_in[14];

  const int B = in_sizes[0] / (T_STEPS * IN_DIM);   // 2048
  float* feat = (float*)d_ws;                       // [B, 80] fp32 scratch

  dim3 grid(B / 32, 2);                             // 16-row tile x 2 waves; y = direction
  bilstm_scan_kernel<<<grid, 64, 0, stream>>>(x, Wf1, bf1, Wf2, bf2,
                                              Wb1, bb1, Wb2, bb2, feat);
  fc_head_kernel<<<B / 16, 32, 0, stream>>>(feat, Wfc1, bfc1, Wfc2, bfc2,
                                            Wfc3, bfc3, (float*)d_out);
}
